// RNN_chaotic_85753317032367
// MI455X (gfx1250) — matrix operand order repaired and checked by reading, not compiled, not run on hardware
//
#include <hip/hip_runtime.h>
#include <stdint.h>

// ---------------- problem constants ----------------
#define T_STEPS 2048
#define BB 64          // batch
#define II 64          // input dim
#define HH 512         // hidden
#define OO 32          // readout
#define OCC 8          // ctx readout
#define NWG 8          // persistent workgroups in phase 1
#define HS  64         // H columns per workgroup (HH / NWG)
#define KTOT 576       // HH + II (concatenated K dim)
#define STRIDE1 584    // KTOT + 8 halfs pad (bank-conflict-free, 16B aligned rows)
#define STRIDE2 520    // 512 + 8 halfs pad for phase 2 weights
#define ALPHA_  0.1f
#define NSCALE_ 0.29646133681875246f   // sqrt(2*(1.5/sqrt(512))^2/0.1)

typedef __attribute__((ext_vector_type(16))) __bf16          v16bf;
typedef __attribute__((ext_vector_type(8)))  float           v8f;
typedef __attribute__((ext_vector_type(8)))  unsigned short  v8u;
typedef __attribute__((ext_vector_type(16))) unsigned short  v16u;
typedef __attribute__((ext_vector_type(4)))  int             v4i;

typedef __attribute__((address_space(1))) v4i* g4p;   // global 16B-vector ptr
typedef __attribute__((address_space(3))) v4i* l4p;   // LDS 16B-vector ptr

// float -> bf16, round to nearest even
__device__ __forceinline__ unsigned short f2bf(float f) {
  unsigned u = __builtin_bit_cast(unsigned, f);
  unsigned r = u + 0x7fffu + ((u >> 16) & 1u);
  return (unsigned short)(r >> 16);
}

// assemble a 16xbf16 WMMA operand from two 16B chunks
__device__ __forceinline__ v16bf pack_ab(v8u lo, v8u hi) {
  v16u c = __builtin_shufflevector(lo, hi, 0,1,2,3,4,5,6,7,8,9,10,11,12,13,14,15);
  return __builtin_bit_cast(v16bf, c);
}

// 16B global -> LDS async copy (CDNA5 GLOBAL_LOAD_ASYNC_TO_LDS_B128), with fallback
__device__ __forceinline__ void async_cp16(const unsigned short* src, unsigned short* lds) {
#if __has_builtin(__builtin_amdgcn_global_load_async_to_lds_b128)
  __builtin_amdgcn_global_load_async_to_lds_b128(
      (g4p)(uintptr_t)src,                    // global source (AS1)
      (l4p)(uint32_t)(uintptr_t)lds,          // LDS dest: low 32 bits of generic
      0, 0);                                  // offset, cpol
#else
  *(v8u*)lds = *(const v8u*)src;
#endif
}

__device__ __forceinline__ void async_wait0() {
#if __has_builtin(__builtin_amdgcn_s_wait_asynccnt)
  __builtin_amdgcn_s_wait_asynccnt(0);
#elif __has_builtin(__builtin_amdgcn_global_load_async_to_lds_b128)
  asm volatile("s_wait_asynccnt 0" ::: "memory");
#endif
}

// agent-scope multi-workgroup barrier (counter per step) + cluster barrier
// (s_barrier_signal/-3 + s_barrier_wait/-3 are architectural NOPs if the
//  dispatch is not a cluster, so this is safe under a plain launch).
__device__ __forceinline__ void grid_barrier(int* cnt, int slot, int tid) {
  __builtin_amdgcn_fence(__ATOMIC_RELEASE, "agent");
  __syncthreads();
  if (tid == 0) {
    __hip_atomic_fetch_add(&cnt[slot], 1, __ATOMIC_RELEASE, __HIP_MEMORY_SCOPE_AGENT);
    while (__hip_atomic_load(&cnt[slot], __ATOMIC_ACQUIRE, __HIP_MEMORY_SCOPE_AGENT) < NWG)
      __builtin_amdgcn_s_sleep(1);
  }
  __builtin_amdgcn_s_cluster_barrier();
  __syncthreads();
  __builtin_amdgcn_fence(__ATOMIC_ACQUIRE, "agent");
}

// =====================================================================
// Phase 1: persistent recurrent loop. 8 WGs, each owns 64 H-columns.
// Per step each wave computes two 16x16 tiles of rate_new via
// v_wmma_f32_16x16x32_bf16 over K=576 (18 k-steps).
// =====================================================================
__global__ __launch_bounds__(256) void rnn_phase1(
    const float* __restrict__ x, const float* __restrict__ rate0,
    const float* __restrict__ noise, const float* __restrict__ Wi,
    const float* __restrict__ bi, const float* __restrict__ Wh,
    const float* __restrict__ bh, float* __restrict__ rate_all,
    unsigned short* __restrict__ xchg0, unsigned short* __restrict__ xchg1,
    int* __restrict__ cnt)
{
  __shared__ __align__(16) unsigned short sW[HS][STRIDE1];   // [h_local][K] bf16
  __shared__ __align__(16) unsigned short sAct[BB][STRIDE1]; // [b][K]=rate|x bf16
  __shared__ float sBias[HS];

  const int tid  = threadIdx.x;
  const int wg   = blockIdx.x;
  const int lane = tid & 31, wave = tid >> 5;
  const int lm = lane & 15, lg = lane >> 4;
  const int mt  = wave >> 1;          // batch-row tile 0..3
  const int nt0 = (wave & 1) * 2;     // first of two H-col tiles

  // ---- preload weight slice (Wh slice | Wi slice) as bf16 ----
  for (int e = tid; e < HS * KTOT; e += 256) {
    int r = e / KTOT, c = e - r * KTOT;
    float w = (c < HH) ? Wh[(size_t)(wg * HS + r) * HH + c]
                       : Wi[(size_t)(wg * HS + r) * II + (c - HH)];
    sW[r][c] = f2bf(w);
  }
  if (tid < HS) sBias[tid] = bi[wg * HS + tid] + bh[wg * HS + tid];

  // ---- seed exchange buffer (parity 0) with rate0 slice ----
  for (int e = tid; e < BB * HS; e += 256) {
    int r = e >> 6, c = e & 63;
    int h = wg * HS + c;
    xchg0[r * HH + h] = f2bf(rate0[r * HH + h]);
  }

  // ---- persistent previous-rate registers (C/D tile layout) ----
  const int hl0 = nt0 * 16 + lm, hl1 = (nt0 + 1) * 16 + lm;
  const int hg0 = wg * HS + hl0, hg1 = wg * HS + hl1;
  float rp0[8], rp1[8];
#pragma unroll
  for (int r = 0; r < 8; ++r) {
    int b = mt * 16 + r + 8 * lg;
    rp0[r] = rate0[b * HH + hg0];
    rp1[r] = rate0[b * HH + hg1];
  }

  grid_barrier(cnt, 0, tid);

  const int arow = mt * 16 + lm;

  for (int t = 0; t < T_STEPS; ++t) {
    const unsigned short* xr = (t & 1) ? xchg1 : xchg0;
    unsigned short*       xw = (t & 1) ? xchg0 : xchg1;

    // ---- fill sAct rate region [64x512] via async global->LDS ----
#pragma unroll
    for (int j = 0; j < 16; ++j) {
      int cc = tid + 256 * j;           // chunk id, 8 halfs each
      int r = cc >> 6, c8 = cc & 63;
      async_cp16(xr + r * HH + c8 * 8, &sAct[r][c8 * 8]);
    }
    // ---- fill x_t region [64x64] with f32->bf16 convert ----
    {
      const float* xt = x + (size_t)t * BB * II;
      int r = tid >> 2, seg = (tid & 3) * 16;
      const float4* p = (const float4*)(xt + r * II + seg);
      float4 a0 = p[0], a1 = p[1], a2 = p[2], a3 = p[3];
      v8u o0 = { f2bf(a0.x), f2bf(a0.y), f2bf(a0.z), f2bf(a0.w),
                 f2bf(a1.x), f2bf(a1.y), f2bf(a1.z), f2bf(a1.w) };
      v8u o1 = { f2bf(a2.x), f2bf(a2.y), f2bf(a2.z), f2bf(a2.w),
                 f2bf(a3.x), f2bf(a3.y), f2bf(a3.z), f2bf(a3.w) };
      *(v8u*)&sAct[r][HH + seg]     = o0;
      *(v8u*)&sAct[r][HH + seg + 8] = o1;
    }
    async_wait0();
    __syncthreads();

    // ---- GEMM: D[64, hslice] += Act[64,576] x W^T ----
    v8f acc0 = {}; v8f acc1 = {};
#pragma unroll
    for (int ko = 0; ko < KTOT / 32; ++ko) {
      const int kw = 32 * ko;
      // A operand: lanes 0-15 K{0..7,16..23}, lanes 16-31 K{8..15,24..31}
      v16bf a = pack_ab(*(const v8u*)&sAct[arow][kw + 8 * lg],
                        *(const v8u*)&sAct[arow][kw + 16 + 8 * lg]);
      // B operand: lanes 0-15 K{0..7,16..23}, lanes 16-31 K{8..15,24..31}
      v16bf b0 = pack_ab(*(const v8u*)&sW[nt0 * 16 + lm][kw + 8 * lg],
                         *(const v8u*)&sW[nt0 * 16 + lm][kw + 16 + 8 * lg]);
      v16bf b1 = pack_ab(*(const v8u*)&sW[(nt0 + 1) * 16 + lm][kw + 8 * lg],
                         *(const v8u*)&sW[(nt0 + 1) * 16 + lm][kw + 16 + 8 * lg]);
      acc0 = __builtin_amdgcn_wmma_f32_16x16x32_bf16(false, a, false, b0, (short)0, acc0, false, false);
      acc1 = __builtin_amdgcn_wmma_f32_16x16x32_bf16(false, a, false, b1, (short)0, acc1, false, false);
    }

    // ---- epilogue: noise, bias, tanh, leaky update, stores ----
    float*       rall_t = rate_all + (size_t)t * BB * HH;
    const float* nz_t   = noise    + (size_t)t * BB * HH;
#pragma unroll
    for (int r = 0; r < 8; ++r) {
      const int b = mt * 16 + r + 8 * lg;
      {
        float pre = acc0[r] + sBias[hl0] + NSCALE_ * nz_t[b * HH + hg0];
        float rn  = (1.0f - ALPHA_) * rp0[r] + ALPHA_ * tanhf(pre);
        rp0[r] = rn;
        rall_t[b * HH + hg0] = rn;
        xw[b * HH + hg0] = f2bf(rn);
      }
      {
        float pre = acc1[r] + sBias[hl1] + NSCALE_ * nz_t[b * HH + hg1];
        float rn  = (1.0f - ALPHA_) * rp1[r] + ALPHA_ * tanhf(pre);
        rp1[r] = rn;
        rall_t[b * HH + hg1] = rn;
        xw[b * HH + hg1] = f2bf(rn);
      }
    }
    if (t + 1 < T_STEPS)  // warm L2 for next step's noise slice
      __builtin_prefetch(nz_t + BB * HH + (mt * 16 + 8 * lg) * HH + hg0, 0, 0);

    grid_barrier(cnt, t + 1, tid);
  }
}

// =====================================================================
// Phase 2: readouts. [131072,512] x [512,48] (Wo | Woc | zero pad),
// bf16 WMMA, weights in LDS, A converted on the fly. HBM-bound.
// =====================================================================
__global__ __launch_bounds__(256) void rnn_phase2(
    const float* __restrict__ rall, const float* __restrict__ Wo,
    const float* __restrict__ bo, const float* __restrict__ Woc,
    const float* __restrict__ boc, float* __restrict__ out,
    float* __restrict__ outc)
{
  __shared__ __align__(16) unsigned short sB[48][STRIDE2];

  const int tid  = threadIdx.x;
  const int lane = tid & 31, wave = tid >> 5;
  const int lm = lane & 15, lg = lane >> 4;

  for (int e = tid; e < 48 * 512; e += 256) {
    int r = e >> 9, c = e & 511;
    float w = 0.0f;
    if (r < 32)      w = Wo[r * 512 + c];
    else if (r < 40) w = Woc[(r - 32) * 512 + c];
    sB[r][c] = f2bf(w);
  }
  __syncthreads();

  const size_t rowbase = (size_t)blockIdx.x * 256;
  v8f acc[2][3] = {};

#pragma unroll
  for (int ko = 0; ko < 16; ++ko) {
    const int kw = 32 * ko;
    const int k0 = kw + 8 * lg;
    v16bf bmat[3];
#pragma unroll
    for (int n = 0; n < 3; ++n)
      bmat[n] = pack_ab(*(const v8u*)&sB[n * 16 + lm][kw + 8 * lg],
                        *(const v8u*)&sB[n * 16 + lm][kw + 16 + 8 * lg]);
#pragma unroll
    for (int m = 0; m < 2; ++m) {
      size_t grow = rowbase + (size_t)(wave * 2 + m) * 16 + lm;
      const float* ap = rall + grow * 512;
      float4 q0 = *(const float4*)(ap + k0);
      float4 q1 = *(const float4*)(ap + k0 + 4);
      float4 q2 = *(const float4*)(ap + k0 + 16);
      float4 q3 = *(const float4*)(ap + k0 + 20);
      v16u au = { f2bf(q0.x), f2bf(q0.y), f2bf(q0.z), f2bf(q0.w),
                  f2bf(q1.x), f2bf(q1.y), f2bf(q1.z), f2bf(q1.w),
                  f2bf(q2.x), f2bf(q2.y), f2bf(q2.z), f2bf(q2.w),
                  f2bf(q3.x), f2bf(q3.y), f2bf(q3.z), f2bf(q3.w) };
      v16bf a = __builtin_bit_cast(v16bf, au);
#pragma unroll
      for (int n = 0; n < 3; ++n)
        acc[m][n] = __builtin_amdgcn_wmma_f32_16x16x32_bf16(false, a, false, bmat[n], (short)0, acc[m][n], false, false);
    }
  }

#pragma unroll
  for (int m = 0; m < 2; ++m) {
#pragma unroll
    for (int r = 0; r < 8; ++r) {
      size_t grow = rowbase + (size_t)(wave * 2 + m) * 16 + r + 8 * lg;
      out[grow * OO + lm]      = acc[m][0][r] + bo[lm];
      out[grow * OO + 16 + lm] = acc[m][1][r] + bo[16 + lm];
      if (lm < OCC)
        outc[grow * OCC + lm]  = acc[m][2][r] + boc[lm];
    }
  }
}

// =====================================================================
extern "C" void kernel_launch(void* const* d_in, const int* in_sizes, int n_in,
                              void* d_out, int out_size, void* d_ws, size_t ws_size,
                              hipStream_t stream) {
  (void)in_sizes; (void)n_in; (void)out_size; (void)ws_size;
  const float* x     = (const float*)d_in[0];
  const float* rate0 = (const float*)d_in[1];
  const float* noise = (const float*)d_in[2];
  const float* Wi    = (const float*)d_in[3];
  const float* bi    = (const float*)d_in[4];
  const float* Wh    = (const float*)d_in[5];
  const float* bh    = (const float*)d_in[6];
  const float* Wo    = (const float*)d_in[7];
  const float* bo    = (const float*)d_in[8];
  const float* Woc   = (const float*)d_in[9];
  const float* boc   = (const float*)d_in[10];

  float* out  = (float*)d_out;
  float* outc = out  + (size_t)T_STEPS * BB * OO;
  float* rall = outc + (size_t)T_STEPS * BB * OCC;

  int* cnt = (int*)d_ws;                                   // (T+1) barrier slots
  unsigned short* xchg0 = (unsigned short*)((char*)d_ws + 16384);
  unsigned short* xchg1 = xchg0 + BB * HH;                 // double-buffered bf16 rate

  (void)hipMemsetAsync(d_ws, 0, (T_STEPS + 1) * sizeof(int), stream);
  rnn_phase1<<<NWG, 256, 0, stream>>>(x, rate0, noise, Wi, bi, Wh, bh,
                                      rall, xchg0, xchg1, cnt);
  rnn_phase2<<<(T_STEPS * BB) / 256, 256, 0, stream>>>(rall, Wo, bo, Woc, boc, out, outc);
}
